// CellGraphTransformer_65120294142396
// MI455X (gfx1250) — matrix operand order repaired and checked by reading, not compiled, not run on hardware
//
#include <hip/hip_runtime.h>
#include <hip/hip_bf16.h>

// ---------------------------------------------------------------------------
// Problem constants (must match reference)
// ---------------------------------------------------------------------------
#define LNUM 4
#define BB   64
#define NN   511
#define SS   512           // S = N + 1
#define DD   256
#define HH   8
#define HDIM 32            // D / H
#define DFFV 512
#define MTOT (BB * SS)     // 32768 rows

typedef __attribute__((ext_vector_type(16))) __bf16 v16bf;
typedef __attribute__((ext_vector_type(8)))  __bf16 v8bf;
typedef __attribute__((ext_vector_type(8)))  float  v8f;

// fp32 -> bf16, round-to-nearest-even
__device__ __forceinline__ __bf16 f2bf(float f) {
    unsigned u = __builtin_bit_cast(unsigned, f);
    unsigned r = u + 0x7FFFu + ((u >> 16) & 1u);
    unsigned short h = (unsigned short)(r >> 16);
    return __builtin_bit_cast(__bf16, h);
}

// A-fragment loader (16x32 bf16, M x K) for wave32 WMMA.
// Caller passes p = rowBase + k + hg*16 where hg = lane>>4.
// lanes 0-15 (hg=0): elems 0..15 = K[k..k+15]
// lanes 16-31(hg=1): elems 0..15 = K[k+16..k+31]
__device__ __forceinline__ v16bf load_a(const __bf16* p) {
    v8bf lo = *(const v8bf*)(p);
    v8bf hi = *(const v8bf*)(p + 8);
    v16bf r;
#pragma unroll
    for (int i = 0; i < 8; ++i) { r[i] = lo[i]; r[i + 8] = hi[i]; }
    return r;
}

// B-fragment: lane l holds column n = l&15, elements e -> k = kBase + 16*hg + e.
// Contiguous 32B load from a [n][k] (row-major N x K) matrix.
__device__ __forceinline__ v16bf load_b(const __bf16* rowN_base_plus_k) {
    return *(const v16bf*)(rowN_base_plus_k);
}

#define WMMA_BF16(a, b, c) \
    __builtin_amdgcn_wmma_f32_16x16x32_bf16(false, (a), false, (b), (short)0, (c), false, false)

// Wait for ASYNCcnt <= N (async global<->LDS ops complete in order per wave)
#if __has_builtin(__builtin_amdgcn_s_wait_asynccnt)
#define WAIT_ASYNC(n) __builtin_amdgcn_s_wait_asynccnt(n)
#else
#define WAIT_ASYNC(n) asm volatile("s_wait_asynccnt %0" ::"i"(n) : "memory")
#endif

// One per-lane 16-byte async copy global -> LDS (tracked by ASYNCcnt).
// ldsPtr: generic pointer into __shared__ (low 32 bits = wave-relative LDS addr)
__device__ __forceinline__ void async_copy_b128(const void* gptr, void* ldsPtr) {
    unsigned laddr = (unsigned)(unsigned long long)ldsPtr;
    asm volatile("global_load_async_to_lds_b128 %0, %1, off"
                 :: "v"(laddr), "v"(gptr) : "memory");
}

// ---------------------------------------------------------------------------
// fp32 -> bf16 weight conversion
// ---------------------------------------------------------------------------
__global__ void cvt_kernel(const float* __restrict__ src, __bf16* __restrict__ dst, long n) {
    long i = (long)blockIdx.x * blockDim.x + threadIdx.x;
    if (i < n) dst[i] = f2bf(src[i]);
}

// ---------------------------------------------------------------------------
// Embedding: x[b,0,:]=game_token ; x[b,1+n,:]=patch_emb[cell_states[b*N+n]]
// ---------------------------------------------------------------------------
__global__ void embed_kernel(const int* __restrict__ cs, const float* __restrict__ patch,
                             const float* __restrict__ game, float* __restrict__ x) {
    long idx = (long)blockIdx.x * blockDim.x + threadIdx.x;
    if (idx >= (long)BB * SS * DD) return;
    int  d = (int)(idx % DD);
    long r = idx / DD;
    int  s = (int)(r % SS);
    int  b = (int)(r / SS);
    x[idx] = (s == 0) ? game[d] : patch[cs[b * NN + (s - 1)] * DD + d];
}

// ---------------------------------------------------------------------------
// LayerNorm over D=256: one block of 256 threads per row; bf16 output
// ---------------------------------------------------------------------------
__global__ __launch_bounds__(256) void ln_kernel(const float* __restrict__ x,
                                                 const float* __restrict__ g,
                                                 const float* __restrict__ bt,
                                                 __bf16* __restrict__ out) {
    __shared__ float red[256];
    int  m = blockIdx.x;
    int  t = threadIdx.x;
    float v = x[(long)m * DD + t];
    red[t] = v;
    __syncthreads();
#pragma unroll
    for (int s = 128; s > 0; s >>= 1) { if (t < s) red[t] += red[t + s]; __syncthreads(); }
    float mean = red[0] * (1.0f / DD);
    __syncthreads();
    float d = v - mean;
    red[t] = d * d;
    __syncthreads();
#pragma unroll
    for (int s = 128; s > 0; s >>= 1) { if (t < s) red[t] += red[t + s]; __syncthreads(); }
    float var = red[0] * (1.0f / DD);
    float nrm = d * rsqrtf(var + 1e-5f);
    out[(long)m * DD + t] = f2bf(nrm * g[t] + bt[t]);
}

// ---------------------------------------------------------------------------
// Generic bf16 WMMA GEMM:  C(M,N) = A(M,K) @ W(N,K)^T + bias
//   mode 0: QKV  (N=768)  -> scatter into qb (B,H,S,HD), kb (B,H,S,HD), vb (B,H,HD,S), bf16
//   mode 1: out-proj      -> xres[m,n] += c + bias   (residual add, fp32)
//   mode 2: FF1           -> outb[m,n] = bf16(relu(c + bias))
//   mode 3: FF2 (K=512)   -> xres[m,n] += c + bias
//
// Block = 128 threads (4 waves); block tile 128x64; each wave computes 32x64.
// B tile (64 x 32k bf16, [n][k] layout) is staged in LDS via
// GLOBAL_LOAD_ASYNC_TO_LDS_B128 with double buffering (ASYNCcnt protocol):
//   iter i: issue chunk i+1 into buf^1; s_wait_asynccnt 2 -> chunk i landed
//   (loads complete in order); barrier; ds_read B frags; 8 WMMA; barrier.
// A fragments stream directly from global (unique rows per wave).
// ---------------------------------------------------------------------------
__global__ __launch_bounds__(128) void gemm_kernel(
    const __bf16* __restrict__ A, const __bf16* __restrict__ W,
    const float* __restrict__ bias, int K, int mode,
    float* __restrict__ xres, __bf16* __restrict__ outb,
    __bf16* __restrict__ qb, __bf16* __restrict__ kb, __bf16* __restrict__ vb) {
    __shared__ __bf16 Bt[2][64 * 32];   // double-buffered B tile, 4 KB each

    int t     = threadIdx.x;
    int lane  = t & 31;
    int wv    = t >> 5;
    int hg    = lane >> 4;
    int ln16  = lane & 15;
    int mBase = blockIdx.x * 128 + wv * 32;
    int nBase = blockIdx.y * 64;

    // cooperative async copy of one 64x32 B tile (4 KB) = 2 b128 per thread
    auto issue = [&](int buf, int k) {
#pragma unroll
        for (int p = 0; p < 2; ++p) {
            int idx = (p * 128 + t) * 8;     // element index in tile
            int n   = idx >> 5;              // 0..63
            int kk  = idx & 31;              // 0..31 (multiple of 8)
            const __bf16* g = W + (long)(nBase + n) * K + k + kk;
            async_copy_b128((const void*)g, (void*)&Bt[buf][n * 32 + kk]);
        }
    };

    v8f c[8] = {};
    const __bf16* arow0 = A + (long)(mBase + ln16) * K;
    const __bf16* arow1 = A + (long)(mBase + 16 + ln16) * K;

    int nIter = K >> 5;
    issue(0, 0);
    for (int it = 0; it < nIter; ++it) {
        int k = it << 5;
        if (it + 1 < nIter) { issue((it + 1) & 1, k + 32); WAIT_ASYNC(2); }
        else                { WAIT_ASYNC(0); }
        __syncthreads();

        const __bf16* lb = &Bt[it & 1][0];
        v16bf b0 = load_b(lb + ((0 * 16 + ln16) * 32 + 16 * hg));
        v16bf b1 = load_b(lb + ((1 * 16 + ln16) * 32 + 16 * hg));
        v16bf b2 = load_b(lb + ((2 * 16 + ln16) * 32 + 16 * hg));
        v16bf b3 = load_b(lb + ((3 * 16 + ln16) * 32 + 16 * hg));
        v16bf a0 = load_a(arow0 + k + hg * 16);
        v16bf a1 = load_a(arow1 + k + hg * 16);

        c[0] = WMMA_BF16(a0, b0, c[0]);
        c[1] = WMMA_BF16(a0, b1, c[1]);
        c[2] = WMMA_BF16(a0, b2, c[2]);
        c[3] = WMMA_BF16(a0, b3, c[3]);
        c[4] = WMMA_BF16(a1, b0, c[4]);
        c[5] = WMMA_BF16(a1, b1, c[5]);
        c[6] = WMMA_BF16(a1, b2, c[6]);
        c[7] = WMMA_BF16(a1, b3, c[7]);
        __syncthreads();   // all waves done reading buf before it is re-filled
    }

#pragma unroll
    for (int f = 0; f < 8; ++f) {
        v8f cc = c[f];
        int j    = f & 3;
        int ncol = nBase + j * 16 + ln16;
        int m0   = mBase + (f >> 2) * 16;
        float bs = bias[ncol];
#pragma unroll
        for (int r = 0; r < 8; ++r) {
            int   m   = m0 + r + 8 * hg;
            float val = cc[r] + bs;
            if (mode == 0) {
                int seg    = ncol >> 8;        // 0=q 1=k 2=v
                int within = ncol & 255;
                int head   = within >> 5;
                int hd     = within & 31;
                int bI     = m >> 9;           // S = 512
                int s      = m & 511;
                if (seg == 2) {
                    vb[(((long)(bI * HH + head)) * HDIM + hd) * SS + s] = f2bf(val);
                } else {
                    long qi = (((long)(bI * HH + head)) * SS + s) * HDIM + hd;
                    if (seg == 0) qb[qi] = f2bf(val);
                    else          kb[qi] = f2bf(val);
                }
            } else if (mode == 2) {
                outb[(long)m * DFFV + ncol] = f2bf(fmaxf(val, 0.0f));
            } else { // 1 or 3: residual add into x
                xres[(long)m * DD + ncol] += val;
            }
        }
    }
}

// ---------------------------------------------------------------------------
// Attention: one wave (32 threads) per (b, h, 16-query tile).
//   scores = (q k^T)/sqrt(HD) + edge_bias  -> softmax -> o = attn @ v
// q: (B,H,S,HD) bf16 ; k: (B,H,S,HD) bf16 ; v: (B,H,HD,S) bf16 (transposed)
// o written as (B,S,D) bf16 with feature index h*HD+hd.
// ---------------------------------------------------------------------------
__global__ __launch_bounds__(32) void attn_kernel(
    const __bf16* __restrict__ qb, const __bf16* __restrict__ kb,
    const __bf16* __restrict__ vb, const int* __restrict__ etm,
    const float* __restrict__ eemb, __bf16* __restrict__ ob) {
    __shared__ float  sc[16][SS];   // 32 KB
    __shared__ __bf16 at[16][SS];   // 16 KB
    int lane = threadIdx.x;
    int hg   = lane >> 4;
    int ln16 = lane & 15;
    int sq0  = blockIdx.x * 16;
    int h    = blockIdx.y;
    int b    = blockIdx.z;
    const __bf16* qbase = qb + ((long)(b * HH + h) * SS) * HDIM;
    const __bf16* kbase = kb + ((long)(b * HH + h) * SS) * HDIM;
    const __bf16* vbase = vb + ((long)(b * HH + h) * HDIM) * SS;
    const float rs = 0.17677669529663687f; // 1/sqrt(32)

    // scores: K = HD = 32 -> single k-chunk per 16x16 key tile
    v16bf aq = load_a(qbase + (long)(sq0 + ln16) * HDIM + hg * 16);
    for (int nt = 0; nt < SS / 16; ++nt) {
        int   sk0 = nt * 16;
        v16bf bk  = load_b(kbase + (long)(sk0 + ln16) * HDIM + 16 * hg);
        v8f   c   = {};
        c = WMMA_BF16(aq, bk, c);
        int sk = sk0 + ln16;
#pragma unroll
        for (int r = 0; r < 8; ++r) {
            int   rr = r + 8 * hg;
            int   sq = sq0 + rr;
            float v  = c[r] * rs;
            if (sq > 0 && sk > 0)
                v += eemb[etm[((long)b * NN + (sq - 1)) * NN + (sk - 1)] * HH + h];
            sc[rr][sk] = v;
        }
    }
    __syncthreads();

    // softmax over 512 keys: lane l < 16 owns row l
    if (lane < 16) {
        float mx = -1e30f;
        for (int k2 = 0; k2 < SS; ++k2) mx = fmaxf(mx, sc[lane][k2]);
        float sm = 0.0f;
        for (int k2 = 0; k2 < SS; ++k2) {
            float e = __expf(sc[lane][k2] - mx);
            sm += e;
            sc[lane][k2] = e;
        }
        float inv = 1.0f / sm;
        for (int k2 = 0; k2 < SS; ++k2) at[lane][k2] = f2bf(sc[lane][k2] * inv);
    }
    __syncthreads();

    // o(16 x 32) = attn(16 x 512) @ V(512 x 32)
    v8f o0 = {}, o1 = {};
    for (int kt = 0; kt < SS; kt += 32) {
        v16bf aa = load_a(&at[ln16][0] + kt + hg * 16);
        v16bf b0 = load_b(vbase + (long)(ln16)      * SS + kt + 16 * hg);
        v16bf b1 = load_b(vbase + (long)(16 + ln16) * SS + kt + 16 * hg);
        o0 = WMMA_BF16(aa, b0, o0);
        o1 = WMMA_BF16(aa, b1, o1);
    }
#pragma unroll
    for (int r = 0; r < 8; ++r) {
        int  sq  = sq0 + r + 8 * hg;
        long row = ((long)b * SS + sq) * DD + h * HDIM;
        ob[row + ln16]      = f2bf(o0[r]);
        ob[row + 16 + ln16] = f2bf(o1[r]);
    }
}

// ---------------------------------------------------------------------------
// Head: g = x[:,0,:]; o = relu(g@fc1^T+b); policy = o@pol^T+b; value = tanh(o@val^T+b)
// d_out: policy (64*7) then value (64)
// ---------------------------------------------------------------------------
__global__ __launch_bounds__(64) void head_kernel(
    const float* __restrict__ x, const float* __restrict__ fc1w, const float* __restrict__ fc1b,
    const float* __restrict__ polw, const float* __restrict__ polb,
    const float* __restrict__ valw, const float* __restrict__ valb, float* __restrict__ out) {
    __shared__ float gsh[DD];
    __shared__ float osh[64];
    int b = blockIdx.x, t = threadIdx.x;
    for (int i = t; i < DD; i += 64) gsh[i] = x[((long)b * SS) * DD + i];
    __syncthreads();
    float acc = fc1b[t];
    for (int k = 0; k < DD; ++k) acc += gsh[k] * fc1w[t * DD + k];
    osh[t] = fmaxf(acc, 0.0f);
    __syncthreads();
    if (t < 7) {
        float p = polb[t];
        for (int k = 0; k < 64; ++k) p += osh[k] * polw[t * 64 + k];
        out[b * 7 + t] = p;
    } else if (t == 7) {
        float v = valb[0];
        for (int k = 0; k < 64; ++k) v += osh[k] * valw[k];
        out[BB * 7 + b] = tanhf(v);
    }
}

// ---------------------------------------------------------------------------
// Host launcher
// ---------------------------------------------------------------------------
extern "C" void kernel_launch(void* const* d_in, const int* in_sizes, int n_in,
                              void* d_out, int out_size, void* d_ws, size_t ws_size,
                              hipStream_t stream) {
    const int*   cs    = (const int*)d_in[0];
    const int*   etm   = (const int*)d_in[1];
    const float* patch = (const float*)d_in[2];
    const float* game  = (const float*)d_in[3];
    const float* eemb  = (const float*)d_in[4];
    const float* in_w  = (const float*)d_in[5];
    const float* in_b  = (const float*)d_in[6];
    const float* out_w = (const float*)d_in[7];
    const float* out_b = (const float*)d_in[8];
    const float* ff1_w = (const float*)d_in[9];
    const float* ff1_b = (const float*)d_in[10];
    const float* ff2_w = (const float*)d_in[11];
    const float* ff2_b = (const float*)d_in[12];
    const float* ln1g  = (const float*)d_in[13];
    const float* ln1b  = (const float*)d_in[14];
    const float* ln2g  = (const float*)d_in[15];
    const float* ln2b  = (const float*)d_in[16];
    const float* fc1w  = (const float*)d_in[17];
    const float* fc1b  = (const float*)d_in[18];
    const float* polw  = (const float*)d_in[19];
    const float* polb  = (const float*)d_in[20];
    const float* valw  = (const float*)d_in[21];
    const float* valb  = (const float*)d_in[22];
    (void)in_sizes; (void)n_in; (void)out_size; (void)ws_size;

    char* ws = (char*)d_ws;
    float*  x   = (float*)(ws + 0);            // 32 MB  fp32 activations
    __bf16* hb  = (__bf16*)(ws + 33554432);    // 16 MB  LN output (bf16)
    __bf16* qbf = (__bf16*)(ws + 50331648);    // 16 MB  q (B,H,S,HD)
    __bf16* kbf = (__bf16*)(ws + 67108864);    // 16 MB  k (B,H,S,HD)
    __bf16* vbf = (__bf16*)(ws + 83886080);    // 16 MB  v (B,H,HD,S)
    __bf16* ob  = (__bf16*)(ws + 100663296);   // 16 MB  attention output
    __bf16* ffh = (__bf16*)(ws + 117440512);   // 32 MB  FF hidden (relu, bf16)
    __bf16* w_in  = (__bf16*)(ws + 150994944); // bf16 weights, 4 MB total
    __bf16* w_out = w_in  + (long)LNUM * 3 * DD * DD;
    __bf16* w_f1  = w_out + (long)LNUM * DD * DD;
    __bf16* w_f2  = w_f1  + (long)LNUM * DFFV * DD;

    long n1 = (long)LNUM * 3 * DD * DD;
    long n2 = (long)LNUM * DD * DD;
    long n3 = (long)LNUM * DFFV * DD;
    long n4 = (long)LNUM * DD * DFFV;
    cvt_kernel<<<(int)((n1 + 255) / 256), 256, 0, stream>>>(in_w,  w_in,  n1);
    cvt_kernel<<<(int)((n2 + 255) / 256), 256, 0, stream>>>(out_w, w_out, n2);
    cvt_kernel<<<(int)((n3 + 255) / 256), 256, 0, stream>>>(ff1_w, w_f1,  n3);
    cvt_kernel<<<(int)((n4 + 255) / 256), 256, 0, stream>>>(ff2_w, w_f2,  n4);

    embed_kernel<<<(MTOT * DD) / 256, 256, 0, stream>>>(cs, patch, game, x);

    for (int i = 0; i < LNUM; ++i) {
        ln_kernel<<<MTOT, 256, 0, stream>>>(x, ln1g + i * DD, ln1b + i * DD, hb);
        gemm_kernel<<<dim3(MTOT / 128, 12), 128, 0, stream>>>(
            hb, w_in + (long)i * 3 * DD * DD, in_b + i * 3 * DD,
            DD, 0, nullptr, nullptr, qbf, kbf, vbf);
        attn_kernel<<<dim3(SS / 16, HH, BB), 32, 0, stream>>>(qbf, kbf, vbf, etm, eemb, ob);
        gemm_kernel<<<dim3(MTOT / 128, 4), 128, 0, stream>>>(
            ob, w_out + (long)i * DD * DD, out_b + i * DD,
            DD, 1, x, nullptr, nullptr, nullptr, nullptr);
        ln_kernel<<<MTOT, 256, 0, stream>>>(x, ln2g + i * DD, ln2b + i * DD, hb);
        gemm_kernel<<<dim3(MTOT / 128, 8), 128, 0, stream>>>(
            hb, w_f1 + (long)i * DFFV * DD, ff1_b + i * DFFV,
            DD, 2, nullptr, ffh, nullptr, nullptr, nullptr);
        gemm_kernel<<<dim3(MTOT / 128, 4), 128, 0, stream>>>(
            ffh, w_f2 + (long)i * DD * DFFV, ff2_b + i * DD,
            DFFV, 3, x, nullptr, nullptr, nullptr, nullptr);
    }

    head_kernel<<<BB, 64, 0, stream>>>(x, fc1w, fc1b, polw, polb, valw, valb, (float*)d_out);
}
